// PairFeatureNet_71794673320041
// MI455X (gfx1250) — matrix operand order repaired and checked by reading, not compiled, not run on hardware
//
#include <hip/hip_runtime.h>
#include <math.h>

// ---- problem constants (match reference) ----
#define B_    4
#define N_    512
#define CS_   384
#define CP_   128
#define RELK  32
#define NBIN  65
#define NRBF  32

typedef __attribute__((ext_vector_type(16))) _Float16 v16h;
typedef __attribute__((ext_vector_type(8)))  float    v8f;
typedef __attribute__((ext_vector_type(4)))  float    v4f;

// ExpNormalSmearing constants (cutoff_lower=0, cutoff_upper=5, num_rbf=32)
__device__ __forceinline__ float kExpNegCut() { return 0.006737946999085467f; }          // exp(-5)
__device__ __forceinline__ float kMeanStep()  { return (1.0f - 0.006737946999085467f) / 31.0f; }
__device__ __forceinline__ float kBeta() {
    const float a = (2.0f / 32.0f) * (1.0f - 0.006737946999085467f);
    return 1.0f / (a * a);
}
#define PI_OVER_CUT 0.6283185307179586f   // pi/5
#define CUTF        5.0f

// ============================================================
// Phase 1: Pi[bn,p] = s[bn,:]@W_i + b_i ; Pj likewise. 1 block per (b,n) row.
// Tiny GEMM (0.4 GFLOP total) -> simple LDS-tiled VALU kernel.
// ============================================================
__global__ __launch_bounds__(CP_) void pf_proj_kernel(
    const float* __restrict__ s,
    const float* __restrict__ W_i, const float* __restrict__ b_i,
    const float* __restrict__ W_j, const float* __restrict__ b_j,
    float* __restrict__ Pi, float* __restrict__ Pj)
{
    __shared__ float srow[CS_];
    const int bn = blockIdx.x;          // 0 .. B*N-1
    const int p  = threadIdx.x;         // 0 .. 127
    const float* src = s + (size_t)bn * CS_;
    for (int k = p; k < CS_; k += CP_) srow[k] = src[k];
    __syncthreads();
    float ai = b_i[p];
    float aj = b_j[p];
    #pragma unroll 4
    for (int k = 0; k < CS_; ++k) {
        const float sv = srow[k];
        ai = fmaf(sv, W_i[k * CP_ + p], ai);
        aj = fmaf(sv, W_j[k * CP_ + p], aj);
    }
    Pi[(size_t)bn * CP_ + p] = ai;
    Pj[(size_t)bn * CP_ + p] = aj;
}

// ============================================================
// Phase 2: one block per (b,i) row; 8 waves, wave w owns channels [16w,16w+16).
// GEMM oriented as D = W_t^T (16ch x 32rbf) x rbf^T (32 x 16 pairs):
//   A (resident) = W_t^T chunk, B (per 16-j tile) = rbf features.
// C/D layout => lane owns 8 CONSECUTIVE channels of ONE pair j:
//   vectorized b128 loads (Pj, W_rel) and 2x global_store_b128 NT per tile,
//   relpos bucket / mask computed once per lane per tile.
// ============================================================
__global__ __launch_bounds__(256) void pf_pair_kernel(
    const float* __restrict__ t,     // (B,N,N)
    const float* __restrict__ r,     // (B,N)
    const float* __restrict__ mask,  // (B,N)
    const float* __restrict__ W_rel, // (65,128)
    const float* __restrict__ b_rel, // (128)
    const float* __restrict__ W_t,   // (32,128)
    const float* __restrict__ b_t,   // (128)
    const float* __restrict__ Pi,    // (B*N,128)
    const float* __restrict__ Pj,    // (B*N,128)
    float* __restrict__ out)         // (B,N,N,128)
{
    const int bi     = blockIdx.x;            // b*N + i
    const int lane   = threadIdx.x & 31;
    const int wave   = threadIdx.x >> 5;      // 0..7
    const int pbase  = wave * 16;             // channel chunk base
    const int nsub   = lane & 15;
    const int hi     = lane >> 4;
    const int chbase = pbase + 8 * hi;        // this lane's 8 consecutive channels

    // ---- A-matrix (resident): W_t^T chunk, 16 ch (M) x 32 rbf (K), f16 ----
    // 16-bit A layout: lane row m = nsub (channel pbase+nsub);
    // halves e: K = 8*hi+e (e<8), K = 16+8*hi+(e-8) (e>=8).
    v16h wa;
    #pragma unroll
    for (int e = 0; e < 16; ++e) {
        const int k = (e < 8) ? (8 * hi + e) : (16 + 8 * hi + (e - 8));
        wa[e] = (_Float16)W_t[k * CP_ + pbase + nsub];
    }

    // ---- resident additive per-channel vector: Pi + b_rel + b_t ----
    v4f add0, add1;
    {
        const float* pirow = Pi + (size_t)bi * CP_ + chbase;
        const v4f pi0 = ((const v4f*)pirow)[0];
        const v4f pi1 = ((const v4f*)pirow)[1];
        const v4f br0 = ((const v4f*)(b_rel + chbase))[0];
        const v4f br1 = ((const v4f*)(b_rel + chbase))[1];
        const v4f bt0 = ((const v4f*)(b_t + chbase))[0];
        const v4f bt1 = ((const v4f*)(b_t + chbase))[1];
        add0 = pi0 + br0 + bt0;
        add1 = pi1 + br1 + bt1;
    }

    const float r_i   = r[bi];
    const float m_i   = mask[bi];
    const int   bbase = (bi / N_) * N_;       // b*N
    const float* trow = t + (size_t)bi * N_;

    const float beta = kBeta();
    const float m0   = kExpNegCut();
    const float dm   = kMeanStep();

    for (int j0 = 0; j0 < N_; j0 += 16) {
        const int j = j0 + nsub;              // this lane's pair column

        // ---- B-matrix: rbf^T, 32 (K) x 16 pairs (N), f16 ----
        // 16-bit B layout: lane column n = nsub; halves e: K = 8*hi+e (e<8), K = 16+8*hi+(e-8) (e>=8).
        const float d    = trow[j];
        const float E    = __expf(-d);
        const float cutv = (d < CUTF) ? 0.5f * (__cosf(PI_OVER_CUT * d) + 1.0f) : 0.0f;
        v16h wb;
        #pragma unroll
        for (int e = 0; e < 16; ++e) {
            const int k = (e < 8) ? (8 * hi + e) : (16 + 8 * hi + (e - 8));
            const float diff = E - (m0 + (float)k * dm);
            wb[e] = (_Float16)(cutv * __expf(-beta * diff * diff));
        }

        // ---- W_t^T @ rbf^T : one WMMA, fp32 accumulate ----
        v8f c = {};
        c = __builtin_amdgcn_wmma_f32_16x16x32_f16(
                /*neg_a=*/false, wa, /*neg_b=*/false, wb,
                /*c_mod=*/(short)0, c, /*reuse_a=*/false, /*reuse_b=*/false);

        // ---- per-lane scalars: relpos bucket, pair mask (once per tile) ----
        const float dr     = r_i - r[bbase + j];
        const float bclamp = fminf(fmaxf(rintf(dr), -(float)RELK), (float)RELK);
        const int   bucket = (int)bclamp + RELK;
        const float pm     = m_i * mask[bbase + j];

        // ---- vector epilogue: 4x b128 loads, 2x b128 NT stores ----
        const float* pjrow = Pj + (size_t)(bbase + j) * CP_ + chbase;
        const v4f pj0 = ((const v4f*)pjrow)[0];
        const v4f pj1 = ((const v4f*)pjrow)[1];
        const float* wrrow = W_rel + (size_t)bucket * CP_ + chbase;
        const v4f wr0 = ((const v4f*)wrrow)[0];
        const v4f wr1 = ((const v4f*)wrrow)[1];

        const v4f c0 = __builtin_shufflevector(c, c, 0, 1, 2, 3);
        const v4f c1 = __builtin_shufflevector(c, c, 4, 5, 6, 7);
        const v4f o0 = (c0 + add0 + pj0 + wr0) * pm;
        const v4f o1 = (c1 + add1 + pj1 + wr1) * pm;

        float* orow = out + ((size_t)bi * N_ + j) * CP_ + chbase;
        __builtin_nontemporal_store(o0, (v4f*)orow);
        __builtin_nontemporal_store(o1, (v4f*)orow + 1);
    }
}

extern "C" void kernel_launch(void* const* d_in, const int* in_sizes, int n_in,
                              void* d_out, int out_size, void* d_ws, size_t ws_size,
                              hipStream_t stream)
{
    const float* s     = (const float*)d_in[0];
    const float* t     = (const float*)d_in[1];
    const float* r     = (const float*)d_in[2];
    const float* mask  = (const float*)d_in[3];
    const float* W_i   = (const float*)d_in[4];
    const float* b_i   = (const float*)d_in[5];
    const float* W_j   = (const float*)d_in[6];
    const float* b_j   = (const float*)d_in[7];
    const float* W_rel = (const float*)d_in[8];
    const float* b_rel = (const float*)d_in[9];
    const float* W_t   = (const float*)d_in[10];
    const float* b_t   = (const float*)d_in[11];
    float* out = (float*)d_out;

    // workspace: Pi then Pj, each B*N*CP floats (1 MB each)
    float* Pi = (float*)d_ws;
    float* Pj = Pi + (size_t)B_ * N_ * CP_;

    pf_proj_kernel<<<B_ * N_, CP_, 0, stream>>>(s, W_i, b_i, W_j, b_j, Pi, Pj);
    pf_pair_kernel<<<B_ * N_, 256, 0, stream>>>(t, r, mask, W_rel, b_rel, W_t, b_t,
                                                Pi, Pj, out);
}
